// MultiHeadAttention_50216757625375
// MI455X (gfx1250) — matrix operand order repaired and checked by reading, not compiled, not run on hardware
//
#include <hip/hip_runtime.h>

// ---------------------------------------------------------------------------
// MHA for MI455X (gfx1250): bf16 WMMA (16x16x32, f32 accum) everywhere,
// flash-attention style fused softmax, V stored transposed for PV matmul,
// cooperative LDS staging of K/V tiles, b128 loads for all A operands.
// B=4, S=2048, E=512, H=8, D=64.
// ---------------------------------------------------------------------------

typedef __attribute__((ext_vector_type(16))) __bf16        v16bf;
typedef __attribute__((ext_vector_type(8)))  float         v8f;
typedef __attribute__((ext_vector_type(8)))  unsigned int  v8u;

struct alignas(16) U4 { unsigned int x, y, z, w; };

union ABReg { v8u u; v16bf h; U4 q[2]; };

__device__ __forceinline__ v8f wmma_bf16(const ABReg& a, const ABReg& b, v8f c) {
  return __builtin_amdgcn_wmma_f32_16x16x32_bf16(
      /*neg_a=*/false, a.h, /*neg_b=*/false, b.h,
      /*c_mod=*/(short)0, c, /*reuse_a=*/false, /*reuse_b=*/false);
}

__device__ __forceinline__ unsigned short f2bf(float f) {
  unsigned int u = __float_as_uint(f);
  unsigned int r = u + 0x7FFFu + ((u >> 16) & 1u);   // round-to-nearest-even
  return (unsigned short)(r >> 16);
}
__device__ __forceinline__ unsigned int packbf(float lo, float hi) {
  return (unsigned int)f2bf(lo) | ((unsigned int)f2bf(hi) << 16);
}

// ---------------------------------------------------------------------------
// Prep: f32 -> bf16 convert, and pack consecutive-row pairs (B-operand layout)
// ---------------------------------------------------------------------------
__global__ void cvt_bf16_kernel(const float* __restrict__ src,
                                unsigned short* __restrict__ dst, int n) {
  int i = blockIdx.x * blockDim.x + threadIdx.x;
  if (i < n) dst[i] = f2bf(src[i]);
}

// dst[p*cols + c] = pack(src[2p][c], src[2p+1][c])  (rows = 2*pairRows)
__global__ void pack_pairs_kernel(const float* __restrict__ src,
                                  unsigned int* __restrict__ dst,
                                  int pairRows, int cols) {
  int i = blockIdx.x * blockDim.x + threadIdx.x;
  int n = pairRows * cols;
  if (i < n) {
    int p = i / cols, c = i - p * cols;
    dst[i] = packbf(src[(2 * p) * cols + c], src[(2 * p + 1) * cols + c]);
  }
}

// ---------------------------------------------------------------------------
// QKV projection: one wave = 16x64 tile of [8192 x 64] per (head, matrix).
// Q,K stored bf16 [B*H][S][D]; V stored transposed bf16 [B*H][D][S].
// ---------------------------------------------------------------------------
__global__ __launch_bounds__(256) void qkv_proj_kernel(
    const U4* __restrict__ x4,                           // x bf16 [8192][512] as U4 [8192][64]
    const unsigned int* __restrict__ wqp,                // [H][256][64] packed pairs
    const unsigned int* __restrict__ wkp,
    const unsigned int* __restrict__ wvp,
    const float* __restrict__ bq, const float* __restrict__ bk,
    const float* __restrict__ bv,
    unsigned short* __restrict__ Qb, unsigned short* __restrict__ Kb,
    unsigned short* __restrict__ Vt) {
  int lane = threadIdx.x & 31;
  int widx = threadIdx.x >> 5;
  int w = blockIdx.x * 8 + widx;          // 12288 waves total
  int mat = w >> 12;                      // 0=Q 1=K 2=V
  int rem = w & 4095;
  int h = rem >> 9;
  int mtile = rem & 511;
  int m0 = mtile * 16;
  int b = m0 >> 11;
  int s0 = m0 & 2047;
  int bh = b * 8 + h;

  const unsigned int* wp = (mat == 0) ? wqp : (mat == 1) ? wkp : wvp;
  const float* bias = (mat == 0) ? bq : (mat == 1) ? bk : bv;

  int hi = lane >> 4, n16 = lane & 15;
  int arow = m0 + n16;                    // A-layout row = lane%16

  v8f acc[4] = {};
  for (int kb = 0; kb < 16; ++kb) {       // E = 512 in 32-chunks
    int e0 = kb * 32;
    ABReg a;                              // A operand: two b128 loads
    a.q[0] = x4[(size_t)arow * 64 + kb * 4 + hi];
    a.q[1] = x4[(size_t)arow * 64 + kb * 4 + hi + 2];
#pragma unroll
    for (int dn = 0; dn < 4; ++dn) {
      ABReg bm;
#pragma unroll
      for (int j = 0; j < 8; ++j) {
        int k = e0 + (j >> 2) * 16 + hi * 8 + 2 * (j & 3);     // B-layout: rows 16*(j/4)+8*hi+2*(j%4), +1
        bm.u[j] = wp[((size_t)h * 256 + (k >> 1)) * 64 + dn * 16 + n16];
      }
      acc[dn] = wmma_bf16(a, bm, acc[dn]);
    }
  }

  if (mat < 2) {
    unsigned short* Out = (mat == 0) ? Qb : Kb;
#pragma unroll
    for (int dn = 0; dn < 4; ++dn) {
      int d = dn * 16 + n16;
      float bs = bias[h * 64 + d];
#pragma unroll
      for (int r = 0; r < 8; ++r) {       // C row M = r + 8*hi
        int s = s0 + hi * 8 + r;
        Out[((size_t)bh * 2048 + s) * 64 + d] = f2bf(acc[dn][r] + bs);
      }
    }
  } else {
#pragma unroll
    for (int dn = 0; dn < 4; ++dn) {
      int d = dn * 16 + n16;
      float bs = bias[h * 64 + d];
      U4 pk;
      pk.x = packbf(acc[dn][0] + bs, acc[dn][1] + bs);
      pk.y = packbf(acc[dn][2] + bs, acc[dn][3] + bs);
      pk.z = packbf(acc[dn][4] + bs, acc[dn][5] + bs);
      pk.w = packbf(acc[dn][6] + bs, acc[dn][7] + bs);
      int s = s0 + hi * 8;                // 8 contiguous s per lane
      *(U4*)&Vt[((size_t)bh * 64 + d) * 2048 + s] = pk;
    }
  }
}

// ---------------------------------------------------------------------------
// Flash attention: block = 8 waves x 16 query rows, all waves share (b,h).
// Per 32-key block: cooperative b128 staging of K(32x64) and Vt(64x32) into
// LDS, 4 WMMAs (scores, B from LDS), online softmax, P transpose via LDS,
// 4 WMMAs (PV, B from LDS).
// ---------------------------------------------------------------------------
__global__ __launch_bounds__(256) void attn_kernel(
    const U4* __restrict__ Q4,             // bf16 [B*H][2048][64] as U4 [..][8]
    const U4* __restrict__ K4,             // bf16 [B*H][2048][64] as U4
    const U4* __restrict__ V4,             // bf16 [B*H][64][2048] as U4 [..][256]
    const int* __restrict__ mask,          // [B][2048][2048]
    unsigned short* __restrict__ Oat) {    // bf16 [B][2048][512]
  __shared__ U4 kbuf[256];                 // 4KB: K rows [32][64] bf16
  __shared__ U4 vbuf[256];                 // 4KB: Vt [64][32] bf16
  __shared__ U4 pbuf[8 * 64];              // 8KB: 16x32 bf16 per wave

  int tid = threadIdx.x;
  int lane = tid & 31;
  int widx = tid >> 5;
  int bh = blockIdx.y;
  int b = bh >> 3, h = bh & 7;
  int qt = blockIdx.x * 128 + widx * 16;   // query row start for this wave
  int hi = lane >> 4, n16 = lane & 15;

  // Load this wave's 16x64 Q tile: two b128 loads per 32-d chunk
  const U4* qg = Q4 + (size_t)bh * 2048 * 8;
  int arow = qt + n16;
  ABReg qa[2];
#pragma unroll
  for (int c = 0; c < 2; ++c) {
    qa[c].q[0] = qg[(size_t)arow * 8 + c * 4 + hi];
    qa[c].q[1] = qg[(size_t)arow * 8 + c * 4 + hi + 2];
  }

  float mrow[8], lrow[8];
#pragma unroll
  for (int r = 0; r < 8; ++r) { mrow[r] = -1e30f; lrow[r] = 0.f; }
  v8f o[4] = {};

  const U4* kg = K4 + (size_t)bh * 2048 * 8;    // row = 8 U4
  const U4* vg = V4 + (size_t)bh * 64 * 256;    // d-row = 256 U4
  size_t mbase = ((size_t)b * 2048 + qt + 8 * hi) * 2048;  // + r*2048 + t
  U4* mypb = &pbuf[widx * 64];
  unsigned short* mypb16 = (unsigned short*)mypb;
  int krow = tid >> 3, kchunk = tid & 7;   // K staging map
  int vd = tid >> 2, vseg = tid & 3;       // V staging map

  for (int kt = 0; kt < 2048; kt += 32) {
    // ---- cooperative staging: K block + Vt block -> LDS (b128 each) ----
    __syncthreads();                       // prior iteration done with LDS
    kbuf[tid] = kg[(size_t)(kt + krow) * 8 + kchunk];
    vbuf[tid] = vg[(size_t)vd * 256 + (kt >> 3) + vseg];
    // prefetch next key block into L2 while we compute on this one
    if (kt + 32 < 2048) {
      __builtin_prefetch(&kg[(size_t)(kt + 32 + krow) * 8 + kchunk], 0, 0);
      __builtin_prefetch(&vg[(size_t)vd * 256 + ((kt + 32) >> 3) + vseg], 0, 0);
    }
    __syncthreads();                       // staged tiles visible

    // ---- scores: two 16x16 tiles (keys kt..+15, +16..+31), K-dim D=64 ----
    v8f sc[2] = {};
#pragma unroll
    for (int tt = 0; tt < 2; ++tt) {
#pragma unroll
      for (int c = 0; c < 2; ++c) {
        ABReg kb;                          // B operand from LDS: two b128
        int base = (tt * 16 + n16) * 8 + c * 4 + hi;
        kb.q[0] = kbuf[base];
        kb.q[1] = kbuf[base + 2];
        sc[tt] = wmma_bf16(qa[c], kb, sc[tt]);
      }
    }
    // ---- scale 1/sqrt(64) + mask (ref: where(mask==0, -1e9)) ----
#pragma unroll
    for (int tt = 0; tt < 2; ++tt)
#pragma unroll
      for (int r = 0; r < 8; ++r) {
        float v = sc[tt][r] * 0.125f;
        int mv = mask[mbase + (size_t)r * 2048 + kt + tt * 16 + n16];
        sc[tt][r] = (mv == 0) ? -1e9f : v;
      }
    // ---- online softmax update ----
    float alpha[8];
#pragma unroll
    for (int r = 0; r < 8; ++r) {
      float mx = fmaxf(sc[0][r], sc[1][r]);
#pragma unroll
      for (int m_ = 1; m_ < 16; m_ <<= 1) mx = fmaxf(mx, __shfl_xor(mx, m_, 32));
      float mnew = fmaxf(mrow[r], mx);
      float a_ = __expf(mrow[r] - mnew);
      float p0 = __expf(sc[0][r] - mnew);
      float p1 = __expf(sc[1][r] - mnew);
      sc[0][r] = p0; sc[1][r] = p1;
      float ps = p0 + p1;
#pragma unroll
      for (int m_ = 1; m_ < 16; m_ <<= 1) ps += __shfl_xor(ps, m_, 32);
      lrow[r] = lrow[r] * a_ + ps;
      mrow[r] = mnew;
      alpha[r] = a_;
    }
#pragma unroll
    for (int dn = 0; dn < 4; ++dn)
#pragma unroll
      for (int r = 0; r < 8; ++r) o[dn][r] *= alpha[r];

    // ---- P: C-layout -> A-layout through LDS (row-major 16x32 bf16) ----
#pragma unroll
    for (int tt = 0; tt < 2; ++tt)
#pragma unroll
      for (int r = 0; r < 8; ++r)
        mypb16[(r + 8 * hi) * 32 + tt * 16 + n16] = f2bf(sc[tt][r]);
    __syncthreads();                       // P stores visible (wave-local use)
    ABReg pa;                              // A operand from LDS: two b128
    pa.q[0] = mypb[n16 * 4 + hi];
    pa.q[1] = mypb[n16 * 4 + hi + 2];
    // ---- PV: O(16x64) += P(16x32) x V(32x64), B from LDS ----
#pragma unroll
    for (int dn = 0; dn < 4; ++dn) {
      ABReg vb;
      int d = dn * 16 + n16;
      vb.q[0] = vbuf[d * 4 + hi];
      vb.q[1] = vbuf[d * 4 + hi + 2];
      o[dn] = wmma_bf16(pa, vb, o[dn]);
    }
  }

  // ---- normalize and store concat-head output bf16 [B][S][H*D] ----
#pragma unroll
  for (int dn = 0; dn < 4; ++dn)
#pragma unroll
    for (int r = 0; r < 8; ++r) {
      float ov = o[dn][r] / lrow[r];
      int s = qt + 8 * hi + r;
      int c = h * 64 + dn * 16 + n16;
      Oat[((size_t)b * 2048 + s) * 512 + c] = f2bf(ov);
    }
}

// ---------------------------------------------------------------------------
// Output projection: out[8192][512] = Oat(bf16) @ Wo + bo, f32 result.
// One wave = 16x64 tile; 4096 waves.
// ---------------------------------------------------------------------------
__global__ __launch_bounds__(256) void oproj_kernel(
    const U4* __restrict__ Oat4,            // bf16 [8192][512] as U4 [8192][64]
    const unsigned int* __restrict__ wop,   // packed pairs [256][512]
    const float* __restrict__ bo, float* __restrict__ out) {
  int lane = threadIdx.x & 31;
  int widx = threadIdx.x >> 5;
  int w = blockIdx.x * 8 + widx;
  int mtile = w >> 3, nblk = w & 7;
  int m0 = mtile * 16, e0 = nblk * 64;
  int hi = lane >> 4, n16 = lane & 15;
  int arow = m0 + n16;

  v8f acc[4] = {};
  for (int kb = 0; kb < 16; ++kb) {
    ABReg a;                               // A operand: two b128 loads
    a.q[0] = Oat4[(size_t)arow * 64 + kb * 4 + hi];
    a.q[1] = Oat4[(size_t)arow * 64 + kb * 4 + hi + 2];
#pragma unroll
    for (int dn = 0; dn < 4; ++dn) {
      ABReg bm;
#pragma unroll
      for (int j = 0; j < 8; ++j) {
        int k = kb * 32 + (j >> 2) * 16 + hi * 8 + 2 * (j & 3);
        bm.u[j] = wop[(size_t)(k >> 1) * 512 + e0 + dn * 16 + n16];
      }
      acc[dn] = wmma_bf16(a, bm, acc[dn]);
    }
  }
#pragma unroll
  for (int dn = 0; dn < 4; ++dn) {
    int col = e0 + dn * 16 + n16;
    float bv = bo[col];
#pragma unroll
    for (int r = 0; r < 8; ++r) {
      int m = m0 + 8 * hi + r;
      out[(size_t)m * 512 + col] = acc[dn][r] + bv;
    }
  }
}

// ---------------------------------------------------------------------------
extern "C" void kernel_launch(void* const* d_in, const int* in_sizes, int n_in,
                              void* d_out, int out_size, void* d_ws, size_t ws_size,
                              hipStream_t stream) {
  const float* x  = (const float*)d_in[0];
  const int* mask = (const int*)d_in[1];
  const float* Wq = (const float*)d_in[2];
  const float* bq = (const float*)d_in[3];
  const float* Wk = (const float*)d_in[4];
  const float* bk = (const float*)d_in[5];
  const float* Wv = (const float*)d_in[6];
  const float* bv = (const float*)d_in[7];
  const float* Wo = (const float*)d_in[8];
  const float* bo = (const float*)d_in[9];
  float* out = (float*)d_out;

  char* ws = (char*)d_ws;
  const size_t MB = 1u << 20;
  unsigned short* xbf = (unsigned short*)(ws);             // 8 MB
  unsigned short* Qb  = (unsigned short*)(ws + 8 * MB);    // 8 MB
  unsigned short* Kb  = (unsigned short*)(ws + 16 * MB);   // 8 MB
  unsigned short* Vt  = (unsigned short*)(ws + 24 * MB);   // 8 MB
  unsigned short* Oat = (unsigned short*)(ws + 32 * MB);   // 8 MB
  unsigned int* wqp = (unsigned int*)(ws + 40 * MB);             // 512 KB each
  unsigned int* wkp = (unsigned int*)(ws + 40 * MB + 512 * 1024);
  unsigned int* wvp = (unsigned int*)(ws + 41 * MB);
  unsigned int* wop = (unsigned int*)(ws + 41 * MB + 512 * 1024);

  // Prep: convert x, pack weights into pair-packed bf16 (B-operand friendly)
  cvt_bf16_kernel<<<(4194304 + 255) / 256, 256, 0, stream>>>(x, xbf, 4194304);
  pack_pairs_kernel<<<(131072 + 255) / 256, 256, 0, stream>>>(Wq, wqp, 2048, 64);
  pack_pairs_kernel<<<(131072 + 255) / 256, 256, 0, stream>>>(Wk, wkp, 2048, 64);
  pack_pairs_kernel<<<(131072 + 255) / 256, 256, 0, stream>>>(Wv, wvp, 2048, 64);
  pack_pairs_kernel<<<(131072 + 255) / 256, 256, 0, stream>>>(Wo, wop, 256, 512);

  // QKV projections: 12288 waves = 1536 blocks x 8 waves
  qkv_proj_kernel<<<1536, 256, 0, stream>>>((const U4*)xbf,
                                            wqp, wkp, wvp, bq, bk, bv,
                                            Qb, Kb, Vt);
  // Flash attention: 16 query super-tiles x 32 (b,h)
  attn_kernel<<<dim3(16, 32), 256, 0, stream>>>((const U4*)Qb, (const U4*)Kb,
                                                (const U4*)Vt, mask, Oat);
  // Output projection: 4096 waves = 512 blocks x 8 waves
  oproj_kernel<<<512, 256, 0, stream>>>((const U4*)Oat, wop, bo, out);
}
